// GraphClassifier_54185307406772
// MI455X (gfx1250) — matrix operand order repaired and checked by reading, not compiled, not run on hardware
//
#include <hip/hip_runtime.h>
#include <hip/hip_bf16.h>
#include <math.h>

#define NUM_GRAPHS 4096
#define LATENT 32

typedef __attribute__((ext_vector_type(16))) __bf16    v16bf;
typedef __attribute__((ext_vector_type(8)))  float     v8f;
typedef __attribute__((ext_vector_type(8)))  unsigned  v8u;

// RNE float->bf16 (used once, for B operand setup)
__device__ __forceinline__ __bf16 f2bf(float f) {
    unsigned u = __builtin_bit_cast(unsigned, f);
    unsigned r = u + 0x7fffu + ((u >> 16) & 1u);
    return __builtin_bit_cast(__bf16, (unsigned short)(r >> 16));
}

// Pack two floats into two bf16 (truncation) with a single v_perm_b32.
__device__ __forceinline__ unsigned pack_bf16(float lo, float hi) {
    return __builtin_amdgcn_perm(__builtin_bit_cast(unsigned, hi),
                                 __builtin_bit_cast(unsigned, lo),
                                 0x07060302u);
}

__device__ __forceinline__ float readlane_f(float v, int l) {
    return __builtin_bit_cast(float,
        __builtin_amdgcn_readlane(__builtin_bit_cast(int, v), l));
}

// Branch-free tanh: 1 - 2/(e^2x + 1) -> v_med3 + v_exp + v_rcp + v_fma
__device__ __forceinline__ float fast_tanh(float x) {
    x = fminf(fmaxf(x, -10.0f), 10.0f);
    float e = __expf(2.0f * x);
    float r = __builtin_amdgcn_rcpf(e + 1.0f);
    return __builtin_fmaf(-2.0f, r, 1.0f);
}

__device__ __forceinline__ float fast_sigmoid(float x) {
    return __builtin_amdgcn_rcpf(1.0f + __expf(-x));
}

// ---------------------------------------------------------------------------
// Kernel 0: zero Asum (4096) + S (4096x32) in workspace.
// ---------------------------------------------------------------------------
__global__ void k_init(float* __restrict__ ws) {
    int i = blockIdx.x * blockDim.x + threadIdx.x;
    if (i < NUM_GRAPHS * (LATENT + 1)) ws[i] = 0.0f;
}

// ---------------------------------------------------------------------------
// Kernel 1: fused attention pass.  One wave = 128 consecutive nodes (8 WMMA
// tiles of 16).  Single pass over z (HBM-bound: ~256 MB @ 23.3 TB/s).
// ---------------------------------------------------------------------------
__global__ void k_attention(const float* __restrict__ z,
                            const int*   __restrict__ batch,
                            const float* __restrict__ W1,   // (32,32) row-major
                            const float* __restrict__ b1v,  // (32,)
                            const float* __restrict__ w2v,  // (32,1)
                            const float* __restrict__ b2v,  // (1,)
                            float* __restrict__ Asum,       // (4096,)
                            float* __restrict__ S,          // (4096,32)
                            int N) {
    const int lane = threadIdx.x & 31;
    const int wave_in_blk = threadIdx.x >> 5;
    const long wave_id = (long)blockIdx.x * (blockDim.x >> 5) + wave_in_blk;
    const long chunk_start = wave_id * 128;
    if (chunk_start >= N) return;

    const int half = lane >> 4;      // 0: lanes 0-15, 1: lanes 16-31
    const int c    = lane & 15;
    const float CSTAB = 6.0f;        // |alpha_raw| <= sqrt(32) ~ 5.66
    const float bias2 = b2v[0] - CSTAB;

    if (chunk_start + 128 <= (long)N) {
        // B operands: B0 = W1[:,0:16], B1 = W1[:,16:32].
        // 16-bit B 32x16 layout: lane c = column; lane half h, element e holds
        // K = 8h + e for e < 8, K = 16+8h+(e-8) for e >= 8 (matches A).
        v16bf B0, B1;
        #pragma unroll
        for (int kk = 0; kk < 16; ++kk) {
            int k = half * 8 + (kk & 7) + ((kk >> 3) << 4);
            B0[kk] = f2bf(W1[k * 32 + c]);
            B1[kk] = f2bf(W1[k * 32 + c + 16]);
        }
        const float b1c0 = b1v[c],  b1c1 = b1v[c + 16];
        const float w2c0 = w2v[c],  w2c1 = w2v[c + 16];

        float acc  = 0.0f;   // per-lane (k = lane) weighted-z accumulator
        float esum = 0.0f;   // per-graph sum of e (uniform across lanes)
        int   curg = batch[chunk_start];

        #pragma unroll 1
        for (int t = 0; t < 8; ++t) {
            const long node0 = chunk_start + (long)t * 16;

            // A operand (16-bit A 16x32 layout): lane (half h, row m=c):
            //   elements 0..7  -> K = 8h + e ; elements 8..15 -> K = 16+8h+(e-8)
            const float4* rp = (const float4*)(z + (node0 + c) * LATENT);
            float4 a0 = rp[2 * half];
            float4 a1 = rp[2 * half + 1];
            float4 a2 = rp[4 + 2 * half];
            float4 a3 = rp[4 + 2 * half + 1];
            v8u Au;
            Au[0] = pack_bf16(a0.x, a0.y);  Au[1] = pack_bf16(a0.z, a0.w);
            Au[2] = pack_bf16(a1.x, a1.y);  Au[3] = pack_bf16(a1.z, a1.w);
            Au[4] = pack_bf16(a2.x, a2.y);  Au[5] = pack_bf16(a2.z, a2.w);
            Au[6] = pack_bf16(a3.x, a3.y);  Au[7] = pack_bf16(a3.z, a3.w);
            v16bf A = __builtin_bit_cast(v16bf, Au);

            v8f C0 = {}, C1 = {};
            C0 = __builtin_amdgcn_wmma_f32_16x16x32_bf16(false, A, false, B0,
                                                         (short)0, C0, false, false);
            C1 = __builtin_amdgcn_wmma_f32_16x16x32_bf16(false, A, false, B1,
                                                         (short)0, C1, false, false);

            // alpha_raw[m] = sum_n tanh(h[m][n]+b1[n]) * w2[n]
            // C layout: VGPR r, lane l: M = r + 8*(l>=16), N = l&15.
            float tr[8];
            #pragma unroll
            for (int r = 0; r < 8; ++r) {
                float v = fast_tanh(C0[r] + b1c0) * w2c0
                        + fast_tanh(C1[r] + b1c1) * w2c1;
                v += __shfl_xor(v, 1, 32);
                v += __shfl_xor(v, 2, 32);
                v += __shfl_xor(v, 4, 32);
                v += __shfl_xor(v, 8, 32);   // all 16 lanes of the half hold sum
                tr[r] = __expf(v + bias2);
            }

            int bl = batch[node0 + c];           // lane m (<16): batch[node0+m]
            int glast = __builtin_amdgcn_readlane(bl, 15);

            if (glast == curg) {
                // Whole tile inside current segment (common: avg 488 nodes/graph)
                #pragma unroll
                for (int m = 0; m < 16; ++m) {
                    float em = readlane_f(tr[m & 7], (m >> 3) * 16);  // SGPR
                    acc  += em * z[(node0 + m) * LATENT + lane];      // L0-hot
                    esum += em;
                }
            } else {
                #pragma unroll
                for (int m = 0; m < 16; ++m) {
                    float em = readlane_f(tr[m & 7], (m >> 3) * 16);
                    int   gm = __builtin_amdgcn_readlane(bl, m);
                    if (gm != curg) {            // uniform branch
                        atomicAdd(&S[curg * LATENT + lane], acc);
                        if (lane == 0) atomicAdd(&Asum[curg], esum);
                        acc = 0.0f; esum = 0.0f; curg = gm;
                    }
                    acc  += em * z[(node0 + m) * LATENT + lane];
                    esum += em;
                }
            }
        }
        atomicAdd(&S[curg * LATENT + lane], acc);
        if (lane == 0) atomicAdd(&Asum[curg], esum);
    } else {
        // scalar tail for N % 128 remainder
        for (long n = chunk_start + lane; n < N; n += 32) {
            const float* zr = z + n * LATENT;
            float alpha = bias2;
            #pragma unroll 1
            for (int j = 0; j < LATENT; ++j) {
                float h = b1v[j];
                #pragma unroll
                for (int k = 0; k < LATENT; ++k) h += zr[k] * W1[k * 32 + j];
                alpha += fast_tanh(h) * w2v[j];
            }
            float e = __expf(alpha);
            int g = batch[n];
            atomicAdd(&Asum[g], e);
            #pragma unroll
            for (int k = 0; k < LATENT; ++k)
                atomicAdd(&S[g * LATENT + k], zr[k] * e);
        }
    }
}

// ---------------------------------------------------------------------------
// Kernel 2: per-graph MLP.  One wave per graph; weights staged in LDS.
// ---------------------------------------------------------------------------
__global__ void k_mlp(const float* __restrict__ Asum,
                      const float* __restrict__ S,
                      const float* __restrict__ m1_w, const float* __restrict__ m1_b,
                      const float* __restrict__ m2_w, const float* __restrict__ m2_b,
                      const float* __restrict__ m3_w, const float* __restrict__ m3_b,
                      float* __restrict__ out) {
    __shared__ float sW1[32 * 128];
    __shared__ float sW2[128 * 64];
    __shared__ float sW3[64];
    __shared__ float sB1[128];
    __shared__ float sB2[64];
    __shared__ float sGZ[8][32];
    __shared__ float sX1[8][128];
    __shared__ float sX2[8][64];

    const int tid  = threadIdx.x;
    const int lane = tid & 31;
    const int w    = tid >> 5;

    for (int i = tid; i < 32 * 128; i += 256) sW1[i] = m1_w[i];
    for (int i = tid; i < 128 * 64; i += 256) sW2[i] = m2_w[i];
    if (tid < 64)  sW3[tid] = m3_w[tid];
    if (tid < 128) sB1[tid] = m1_b[tid];
    if (tid < 64)  sB2[tid] = m2_b[tid];
    __syncthreads();

    const int g = blockIdx.x * 8 + w;
    const float inv = __builtin_amdgcn_rcpf(Asum[g] + 1e-8f);
    sGZ[w][lane] = S[g * LATENT + lane] * inv;
    __syncthreads();

    // layer 1: 32 -> 128, relu
    #pragma unroll
    for (int i = 0; i < 4; ++i) {
        int j = lane + 32 * i;
        float a = sB1[j];
        #pragma unroll
        for (int k = 0; k < 32; ++k) a += sGZ[w][k] * sW1[k * 128 + j];
        sX1[w][j] = fmaxf(a, 0.0f);
    }
    __syncthreads();

    // layer 2: 128 -> 64, relu
    #pragma unroll
    for (int i = 0; i < 2; ++i) {
        int j = lane + 32 * i;
        float a = sB2[j];
        #pragma unroll 8
        for (int k = 0; k < 128; ++k) a += sX1[w][k] * sW2[k * 64 + j];
        sX2[w][j] = fmaxf(a, 0.0f);
    }
    __syncthreads();

    // layer 3: 64 -> 1, sigmoid
    float p = sX2[w][lane] * sW3[lane] + sX2[w][lane + 32] * sW3[lane + 32];
    p += __shfl_xor(p, 1, 32);
    p += __shfl_xor(p, 2, 32);
    p += __shfl_xor(p, 4, 32);
    p += __shfl_xor(p, 8, 32);
    p += __shfl_xor(p, 16, 32);
    if (lane == 0) {
        out[g] = fast_sigmoid(p + m3_b[0]);
    }
}

// ---------------------------------------------------------------------------
extern "C" void kernel_launch(void* const* d_in, const int* in_sizes, int n_in,
                              void* d_out, int out_size, void* d_ws, size_t ws_size,
                              hipStream_t stream) {
    const float* z      = (const float*)d_in[0];
    const int*   batch  = (const int*)  d_in[1];
    const float* att_w1 = (const float*)d_in[2];
    const float* att_b1 = (const float*)d_in[3];
    const float* att_w2 = (const float*)d_in[4];
    const float* att_b2 = (const float*)d_in[5];
    const float* m1_w   = (const float*)d_in[6];
    const float* m1_b   = (const float*)d_in[7];
    const float* m2_w   = (const float*)d_in[8];
    const float* m2_b   = (const float*)d_in[9];
    const float* m3_w   = (const float*)d_in[10];
    const float* m3_b   = (const float*)d_in[11];

    const int N = in_sizes[0] / LATENT;

    float* Asum = (float*)d_ws;              // 4096 floats
    float* S    = Asum + NUM_GRAPHS;         // 4096*32 floats

    {
        int n = NUM_GRAPHS * (LATENT + 1);
        k_init<<<(n + 255) / 256, 256, 0, stream>>>(Asum);
    }
    {
        long waves  = ((long)N + 127) / 128;
        long blocks = (waves + 7) / 8;
        k_attention<<<(int)blocks, 256, 0, stream>>>(
            z, batch, att_w1, att_b1, att_w2, att_b2, Asum, S, N);
    }
    k_mlp<<<NUM_GRAPHS / 8, 256, 0, stream>>>(
        Asum, S, m1_w, m1_b, m2_w, m2_b, m3_w, m3_b, (float*)d_out);
}
